// EFGCNLayer_46273977647230
// MI455X (gfx1250) — matrix operand order repaired and checked by reading, not compiled, not run on hardware
//
#include <hip/hip_runtime.h>
#include <hip/hip_bf16.h>
#include <math.h>

// ---- problem constants (from reference setup_inputs) ----
#define B_   8
#define N_   2048
#define F_   256
#define OUT_ 256
#define KD_  32

typedef __attribute__((ext_vector_type(16))) __bf16       v16bf;
typedef __attribute__((ext_vector_type(2)))  __bf16       v2bf;
typedef __attribute__((ext_vector_type(8)))  float        v8f;
typedef __attribute__((ext_vector_type(8)))  unsigned int v8u;

// 16-bit A-matrix 16x32 layout (ISA 7.12.2): lane m=lane&15, half=lane>>4.
// VGPR v holds K = {base, base+1}, base = (v<4 ? 2v : 16+2(v-4)) + 8*half.
static __device__ __forceinline__ int a_kbase(int v, int hf) {
  return ((v < 4) ? (2 * v) : (16 + 2 * (v - 4))) + 8 * hf;
}
// 16-bit B-matrix 32x16 layout: lane n=lane&15, half=lane>>4.
// VGPR v holds K = {base, base+1}, base = (v<4 ? 2v : 16+2(v-4)) + 8*half, at column n.
static __device__ __forceinline__ int b_kbase(int v, int hf) {
  return ((v < 4) ? (2 * v) : (16 + 2 * (v - 4))) + 8 * hf;
}

// Pack two f32 -> one dword of 2 bf16; vector form lets ISel emit v_cvt_pk_bf16_f32.
static __device__ __forceinline__ unsigned pack2(float a, float b) {
  v2bf p;
  p.x = (__bf16)a;
  p.y = (__bf16)b;
  return __builtin_bit_cast(unsigned, p);
}
static __device__ __forceinline__ v16bf as_v16bf(v8u t) {
  return __builtin_bit_cast(v16bf, t);
}
static __device__ __forceinline__ float sigmoidf_(float x) {
  return 1.0f / (1.0f + __expf(-x));
}

// ---------------------------------------------------------------------------
// Kernel 1: XW = X@W + Wb  -> stored TRANSPOSED bf16:  XWT[b][o][n]  (8 MB)
//           XK = X@K + Kb  -> stored row-major  bf16:  XKb[b][n][k]  (1 MB)
// grid = B*(N/16) blocks of 512 threads (16 waves). Wave w: out cols [16w,16w+16).
// Waves 0,1 additionally produce the two 16-col tiles of XK.
// ---------------------------------------------------------------------------
__global__ void __launch_bounds__(512)
proj_kernel(const float* __restrict__ X,
            const float* __restrict__ W, const float* __restrict__ Wb,
            const float* __restrict__ K, const float* __restrict__ Kb,
            __bf16* __restrict__ XWT, __bf16* __restrict__ XKb) {
  const int lane = threadIdx.x & 31;
  const int w    = threadIdx.x >> 5;
  const int b    = blockIdx.x >> 7;            // N/16 == 128 row-tiles
  const int row0 = (blockIdx.x & 127) << 4;
  const int m    = lane & 15;                  // A row / B col within tile
  const int hf   = lane >> 4;

  // ---- XW tile: 16 rows x 16 cols, K-loop over F in steps of 32 ----
  {
    const int c0 = w << 4;
    v8f acc = {};
    for (int k0 = 0; k0 < F_; k0 += 32) {
      v8u ta, tb;
#pragma unroll
      for (int v = 0; v < 8; ++v) {
        const int ka = k0 + a_kbase(v, hf);
        const float* px = X + ((size_t)(b * N_) + row0 + m) * F_ + ka;
        ta[v] = pack2(px[0], px[1]);           // contiguous K pair
        const int kb = k0 + b_kbase(v, hf);
        const float* pw = W + (size_t)kb * OUT_ + c0 + m;
        tb[v] = pack2(pw[0], pw[OUT_]);        // rows kb, kb+1 of W
      }
      acc = __builtin_amdgcn_wmma_f32_16x16x32_bf16(
          false, as_v16bf(ta), false, as_v16bf(tb), (short)0, acc, false, false);
    }
    const float bias = Wb[c0 + m];
#pragma unroll
    for (int r = 0; r < 8; ++r) {              // D layout: M = r + 8*half, N = lane&15
      const int mr = r + 8 * hf;
      XWT[((size_t)b * OUT_ + c0 + m) * N_ + row0 + mr] = (__bf16)(acc[r] + bias);
    }
  }

  // ---- XK tiles (KD=32): waves 0 and 1 only (wave-uniform branch) ----
  if (w < 2) {
    const int c0 = w << 4;
    v8f acc = {};
    for (int k0 = 0; k0 < F_; k0 += 32) {
      v8u ta, tb;
#pragma unroll
      for (int v = 0; v < 8; ++v) {
        const int ka = k0 + a_kbase(v, hf);
        const float* px = X + ((size_t)(b * N_) + row0 + m) * F_ + ka;
        ta[v] = pack2(px[0], px[1]);
        const int kb = k0 + b_kbase(v, hf);
        const float* pk = K + (size_t)kb * KD_ + c0 + m;
        tb[v] = pack2(pk[0], pk[KD_]);
      }
      acc = __builtin_amdgcn_wmma_f32_16x16x32_bf16(
          false, as_v16bf(ta), false, as_v16bf(tb), (short)0, acc, false, false);
    }
    const float bias = Kb[c0 + m];
#pragma unroll
    for (int r = 0; r < 8; ++r) {
      const int mr = r + 8 * hf;
      XKb[((size_t)(b * N_) + row0 + mr) * KD_ + c0 + m] = (__bf16)(acc[r] + bias);
    }
  }
}

// ---------------------------------------------------------------------------
// Kernel 2: deg[b,i] = 1 + sum_j A[b,i,j]*sigmoid(XK_i . XK_j);  inv = rsqrt(deg)
// One wave per 16-row i-tile; score tile = one bf16 WMMA per 16 j's.
// ---------------------------------------------------------------------------
__global__ void __launch_bounds__(256)
deg_kernel(const __bf16* __restrict__ XKb, const float* __restrict__ A,
           float* __restrict__ invs) {
  const int lane = threadIdx.x & 31;
  const int wg   = blockIdx.x * (blockDim.x >> 5) + (threadIdx.x >> 5);
  const int b    = wg >> 7;
  const int i0   = (wg & 127) << 4;
  const int n    = lane & 15;
  const int hf   = lane >> 4;

  // A operand: XK rows i0..i0+15 (16x32 bf16), loaded once
  v8u ta;
#pragma unroll
  for (int v = 0; v < 8; ++v)
    ta[v] = *(const unsigned*)(XKb + ((size_t)(b * N_) + i0 + n) * KD_ + a_kbase(v, hf));
  const v16bf amat = as_v16bf(ta);

  float rowsum[8];
#pragma unroll
  for (int r = 0; r < 8; ++r) rowsum[r] = 0.0f;

  const float* Ab = A + (size_t)b * N_ * N_;
  for (int j0 = 0; j0 < N_; j0 += 16) {
    v8u tb;
#pragma unroll
    for (int v = 0; v < 8; ++v)   // B[k][n] = XK[j0+n][k]; K pairs contiguous in row
      tb[v] = *(const unsigned*)(XKb + ((size_t)(b * N_) + j0 + n) * KD_ + b_kbase(v, hf));
    v8f s = {};
    s = __builtin_amdgcn_wmma_f32_16x16x32_bf16(
        false, amat, false, as_v16bf(tb), (short)0, s, false, false);
#pragma unroll
    for (int r = 0; r < 8; ++r) {
      const int mr = r + 8 * hf;
      const int j  = j0 + n;
      float val = Ab[(size_t)(i0 + mr) * N_ + j] * sigmoidf_(s[r]);
      if (i0 + mr == j) val += 1.0f;           // + Identity
      rowsum[r] += val;
    }
  }
  // reduce over the 16 lanes of each half (N dimension)
#pragma unroll
  for (int r = 0; r < 8; ++r) {
    float v = rowsum[r];
    for (int off = 1; off < 16; off <<= 1) v += __shfl_xor(v, off, 32);
    if (n == 0) {
      const int mr = r + 8 * hf;
      invs[(size_t)b * N_ + i0 + mr] = (v > 0.0f) ? rsqrtf(v) : 0.0f;
    }
  }
}

// ---------------------------------------------------------------------------
// Kernel 3: out = relu( AI_norm @ XW ), AI recomputed on the fly.
// 16 waves per block own one 16-row i-tile x all 256 out cols.
// Waves 0,1 build the normalized gated 16x32 weight tile in LDS (bf16, laid
// out so it reloads directly as the WMMA A operand). Double-buffered: tile
// t+1 is produced while all 16 waves accumulate tile t -> 1 barrier/iter.
// ---------------------------------------------------------------------------
__global__ void __launch_bounds__(512)
gemm_kernel(const __bf16* __restrict__ XKb, const __bf16* __restrict__ XWT,
            const float* __restrict__ A, const float* __restrict__ invs,
            float* __restrict__ Xout) {
  __shared__ __align__(16) __bf16 Wtile[2][16 * 32];  // [buf][row i][col j]
  const int lane = threadIdx.x & 31;
  const int w    = threadIdx.x >> 5;
  const int b    = blockIdx.x >> 7;
  const int i0   = (blockIdx.x & 127) << 4;
  const int n    = lane & 15;
  const int hf   = lane >> 4;
  const int c0   = w << 4;

  v8u z = {};
  v16bf samat = as_v16bf(z);
  float si[8] = {};
  if (w < 2) {                                  // score-producing waves
    v8u ta;
#pragma unroll
    for (int v = 0; v < 8; ++v)
      ta[v] = *(const unsigned*)(XKb + ((size_t)(b * N_) + i0 + n) * KD_ + a_kbase(v, hf));
    samat = as_v16bf(ta);
#pragma unroll
    for (int r = 0; r < 8; ++r) si[r] = invs[(size_t)b * N_ + i0 + r + 8 * hf];
  }

  const float* Ab = A + (size_t)b * N_ * N_;

  // produce the normalized gated 16x32 weight tile for columns [j0, j0+32)
  auto make_tile = [&](int j0, int buf) {
    const int js = j0 + (w << 4);
    v8u tb;
#pragma unroll
    for (int v = 0; v < 8; ++v)
      tb[v] = *(const unsigned*)(XKb + ((size_t)(b * N_) + js + n) * KD_ + b_kbase(v, hf));
    v8f s = {};
    s = __builtin_amdgcn_wmma_f32_16x16x32_bf16(
        false, samat, false, as_v16bf(tb), (short)0, s, false, false);
    const float sj = invs[(size_t)b * N_ + js + n];
#pragma unroll
    for (int r = 0; r < 8; ++r) {
      const int mr = r + 8 * hf;
      const int j  = js + n;
      float val = Ab[(size_t)(i0 + mr) * N_ + j] * sigmoidf_(s[r]);
      if (i0 + mr == j) val += 1.0f;
      val *= si[r] * sj;                        // D^-1/2 (A*f + I) D^-1/2
      Wtile[buf][mr * 32 + (w << 4) + n] = (__bf16)val;
    }
  };

  v8f acc = {};
  if (w < 2) make_tile(0, 0);                   // prologue
  __syncthreads();

  constexpr int NITER = N_ / 32;                // 64
  for (int t = 0; t < NITER; ++t) {
    if (t + 1 < NITER && w < 2) make_tile((t + 1) * 32, (t + 1) & 1);
    const int j0 = t * 32;
    // A operand straight from LDS (K pairs contiguous, 4B aligned: kbase even)
    v8u ta, tb;
#pragma unroll
    for (int v = 0; v < 8; ++v) {
      ta[v] = *(const unsigned*)(&Wtile[t & 1][n * 32 + a_kbase(v, hf)]);
      // B[k][n] = XW[j0+k][c0+n] = XWT[b][c0+n][j0+k]; K pairs contiguous
      tb[v] = *(const unsigned*)(XWT + ((size_t)b * OUT_ + c0 + n) * N_ + j0 + b_kbase(v, hf));
    }
    acc = __builtin_amdgcn_wmma_f32_16x16x32_bf16(
        false, as_v16bf(ta), false, as_v16bf(tb), (short)0, acc, false, false);
    __syncthreads();
  }

#pragma unroll
  for (int r = 0; r < 8; ++r) {
    const int mr = r + 8 * hf;
    const float v = acc[r];
    Xout[((size_t)(b * N_) + i0 + mr) * OUT_ + c0 + n] = (v > 0.0f) ? v : 0.0f;
  }
}

// ---------------------------------------------------------------------------
extern "C" void kernel_launch(void* const* d_in, const int* in_sizes, int n_in,
                              void* d_out, int out_size, void* d_ws, size_t ws_size,
                              hipStream_t stream) {
  (void)in_sizes; (void)n_in; (void)out_size; (void)ws_size;
  const float* X   = (const float*)d_in[0];
  const float* A   = (const float*)d_in[1];
  const void*  nin = d_in[2];
  const float* W   = (const float*)d_in[3];
  const float* Wb  = (const float*)d_in[4];
  const float* K   = (const float*)d_in[5];
  const float* Kb  = (const float*)d_in[6];

  // workspace: XWT (bf16, B*OUT*N = 8MB) | XKb (bf16, B*N*KD = 1MB) | invs (f32, 64KB)
  char*   ws   = (char*)d_ws;
  __bf16* XWT  = (__bf16*)ws;
  __bf16* XKb  = (__bf16*)(ws + (size_t)B_ * OUT_ * N_ * sizeof(__bf16));
  float*  invs = (float*)(ws + (size_t)B_ * OUT_ * N_ * sizeof(__bf16)
                             + (size_t)B_ * N_ * KD_ * sizeof(__bf16));

  float* Xout = (float*)d_out;
  const size_t xout_elems = (size_t)B_ * N_ * OUT_;
  const size_t a_elems    = (size_t)B_ * N_ * N_;

  const int tiles = B_ * (N_ / 16);             // 1024
  proj_kernel<<<tiles, 512, 0, stream>>>(X, W, Wb, K, Kb, XWT, XKb);
  deg_kernel<<<tiles / 8, 256, 0, stream>>>(XKb, A, invs);
  gemm_kernel<<<tiles, 512, 0, stream>>>(XKb, XWT, A, invs, Xout);

  // pass-through tuple outputs: A and n, concatenated after X_out
  hipMemcpyAsync(Xout + xout_elems, A, a_elems * sizeof(float),
                 hipMemcpyDeviceToDevice, stream);
  hipMemcpyAsync((char*)d_out + (xout_elems + a_elems) * sizeof(float), nin,
                 B_ * sizeof(int), hipMemcpyDeviceToDevice, stream);
}
